// MinkUNet_21294447853814
// MI455X (gfx1250) — matrix operand order repaired and checked by reading, not compiled, not run on hardware
//
#include <hip/hip_runtime.h>

// ---------------------------------------------------------------------------
// MinkUNet-style dense 3D U-Net on MI455X (gfx1250, wave32, WMMA bf16)
// ---------------------------------------------------------------------------

constexpr int   GRIDD  = 64;
constexpr int   GRIDD3 = GRIDD * GRIDD * GRIDD;
constexpr int   OUTC   = 64;
constexpr float VSZ    = 0.05f;

typedef __bf16 bf16_t;
typedef __attribute__((ext_vector_type(8)))  float   v8f;
typedef __attribute__((ext_vector_type(16))) __bf16  v16bf;
typedef __attribute__((ext_vector_type(8)))  __bf16  v8bf;

constexpr int cgcd(int a, int b) { return b == 0 ? a : cgcd(b, a % b); }

// ------------------------------ voxelization -------------------------------

__global__ void scatter_kernel(const float* __restrict__ xyz,
                               const float* __restrict__ feat,
                               float* __restrict__ sums,
                               float* __restrict__ cnt, int n) {
  int p = blockIdx.x * blockDim.x + threadIdx.x;
  if (p >= n) return;
  int ix = (int)floorf(xyz[p * 3 + 0] * (1.0f / VSZ));
  int iy = (int)floorf(xyz[p * 3 + 1] * (1.0f / VSZ));
  int iz = (int)floorf(xyz[p * 3 + 2] * (1.0f / VSZ));
  ix = ix < 0 ? 0 : (ix > GRIDD - 1 ? GRIDD - 1 : ix);
  iy = iy < 0 ? 0 : (iy > GRIDD - 1 ? GRIDD - 1 : iy);
  iz = iz < 0 ? 0 : (iz > GRIDD - 1 ? GRIDD - 1 : iz);
  int fl = (ix * GRIDD + iy) * GRIDD + iz;
  atomicAdd(&cnt[fl], 1.0f);
#pragma unroll
  for (int c = 0; c < 4; ++c) atomicAdd(&sums[fl * 4 + c], feat[p * 4 + c]);
}

__global__ void meanify_kernel(const float* __restrict__ sums,
                               const float* __restrict__ cnt,
                               bf16_t* __restrict__ v0) {
  int v = blockIdx.x * blockDim.x + threadIdx.x;
  if (v >= GRIDD3) return;
  float c = cnt[v];
  c = c > 1.0f ? c : 1.0f;
  float inv = 1.0f / c;
#pragma unroll
  for (int ch = 0; ch < 4; ++ch)
    v0[v * 4 + ch] = (bf16_t)(sums[v * 4 + ch] * inv);
}

// --------------------------- weight repack (bf16) --------------------------
// in : fp32 [k^3, Ci, Co] (Co fastest)   out: bf16 [Co][Kpad], zero padded

__global__ void pack_w_kernel(const float* __restrict__ W,
                              bf16_t* __restrict__ Wp,
                              int Ktot, int Kpad, int Co) {
  int t = blockIdx.x * blockDim.x + threadIdx.x;
  if (t >= Co * Kpad) return;
  int n = t / Kpad;
  int k = t - n * Kpad;
  Wp[t] = (k < Ktot) ? (bf16_t)W[(size_t)k * Co + n] : (bf16_t)0.0f;
}

// --------------------------- implicit-GEMM conv3d --------------------------
// One wave32 = 16 voxels x (16*NT) Cout. K = k^3*Ci consumed 32 per
// v_wmma_f32_16x16x32_bf16, fp32 accumulate. Activations are channel-last
// and Ci % 16 == 0 (fast path), so every 16-K WMMA half lies inside one
// spatial tap with contiguous ci -> A fragments come straight from global
// memory. Tap decomposition/bounds/address are hoisted: a "group" of
// lcm(32,Ci) K covers GT taps and GC chunks with compile-time tap selection.

template <int KS, bool DECONV, int CI>
__device__ __forceinline__ void tap_addr(const bf16_t* __restrict__ in,
                                         int tap, int ox, int oy, int oz,
                                         int Din, int stride, int padlo,
                                         const bf16_t*& base, bool& okt) {
  int kw = tap % KS;
  int r = tap / KS;
  int kh = r % KS;
  int kd = r / KS;
  int ix, iy, iz;
  bool ok;
  if (DECONV) {
    // lhs_dilation=2, pad_lo=2: tap contributes only if dilated coord even
    int jx = ox + kd - padlo, jy = oy + kh - padlo, jz = oz + kw - padlo;
    int orr = jx | jy | jz;
    ok = (orr >= 0) && ((orr & 1) == 0);
    ix = jx >> 1; iy = jy >> 1; iz = jz >> 1;
  } else {
    ix = ox * stride + kd - padlo;
    iy = oy * stride + kh - padlo;
    iz = oz * stride + kw - padlo;
    ok = ((ix | iy | iz) >= 0);
  }
  ok = ok && (ix < Din) && (iy < Din) && (iz < Din);
  okt = ok;
  base = in + (((size_t)ix * Din + iy) * Din + iz) * CI;
}

template <int KS, bool DECONV, int CI>
__device__ __forceinline__ v8bf load_a_frag(const bf16_t* __restrict__ in,
                                            int kb, int ox, int oy, int oz,
                                            int Din, int stride, int padlo) {
  constexpr int Ktot = KS * KS * KS * CI;
  v8bf v = {};
  int tap = kb / CI;  // CI compile-time: folds to mul/shift
  int ci = kb - tap * CI;
  const bf16_t* base;
  bool ok;
  tap_addr<KS, DECONV, CI>(in, tap, ox, oy, oz, Din, stride, padlo, base, ok);
  ok = ok && (kb < Ktot);
  if (ok) v = *(const v8bf*)(base + ci);
  return v;
}

template <int KS, bool DECONV, int CI, int NT>
__global__ __launch_bounds__(32) void conv3d_wmma_kernel(
    const bf16_t* __restrict__ in, const bf16_t* __restrict__ wp,
    const float* __restrict__ gamma, const float* __restrict__ beta,
    const bf16_t* __restrict__ res, bf16_t* __restrict__ out,
    int Din, int Dout, int Co, int Kpad, int stride, int padlo, int relu) {
  constexpr int Ktot = KS * KS * KS * CI;
  const int lane = threadIdx.x;
  const int m = lane & 15;      // fragment row / B column
  const int hf = lane >> 4;
  const int vbase = blockIdx.x * 16;
  const int nb = blockIdx.y * (16 * NT);

  v8f acc[NT] = {};
  const bf16_t* brow0 = wp + (size_t)(nb + m) * Kpad + hf * 8;

  if constexpr (CI >= 8) {
    // ---------------- fast path: direct global A fragments ----------------
    const int ov = vbase + m;
    const int oz = ov % Dout;
    const int oy = (ov / Dout) % Dout;
    const int ox = ov / (Dout * Dout);

    constexpr int GCD = cgcd(32, CI);
    constexpr int GT = 32 / GCD;        // taps per group
    constexpr int GC = CI / GCD;        // 32-K chunks per group
    constexpr int GK = GT * CI;         // K per group (== GC*32)
    const int ng = (KS * KS * KS) / GT; // full groups

    for (int g = 0; g < ng; ++g) {
      const bf16_t* base[GT];
      bool okt[GT];
#pragma unroll
      for (int tt = 0; tt < GT; ++tt)
        tap_addr<KS, DECONV, CI>(in, g * GT + tt, ox, oy, oz, Din, stride,
                                 padlo, base[tt], okt[tt]);
      if (g + 1 < ng) __builtin_prefetch(brow0 + (size_t)(g + 1) * GK, 0, 3);
#pragma unroll
      for (int c = 0; c < GC; ++c) {
        const int k0 = g * GK + c * 32;
        const int t0 = (c * 32) / CI;        // compile-time after unroll
        const int t1 = (c * 32 + 16) / CI;
        const int o0 = c * 32 - t0 * CI;
        const int o1 = c * 32 + 16 - t1 * CI;
        v8bf a0 = {}, a1 = {};
        if (okt[t0]) a0 = *(const v8bf*)(base[t0] + o0 + hf * 8);
        if (okt[t1]) a1 = *(const v8bf*)(base[t1] + o1 + hf * 8);
        v16bf a;
#pragma unroll
        for (int i = 0; i < 8; ++i) { a[i] = a0[i]; a[i + 8] = a1[i]; }
#pragma unroll
        for (int t = 0; t < NT; ++t) {
          const bf16_t* bp = brow0 + (size_t)t * 16 * Kpad + k0;
          v8bf b0 = *(const v8bf*)bp;
          v8bf b1 = *(const v8bf*)(bp + 16);
          v16bf b;
#pragma unroll
          for (int i = 0; i < 8; ++i) { b[i] = b0[i]; b[i + 8] = b1[i]; }
          acc[t] = __builtin_amdgcn_wmma_f32_16x16x32_bf16(
              false, a, false, b, (short)0, acc[t], false, false);
        }
      }
    }
    // ---- tail: leftover taps + zero-padded K region (generic chunks) ----
    for (int k0 = ng * GK; k0 < Kpad; k0 += 32) {
      v8bf a0 = load_a_frag<KS, DECONV, CI>(in, k0 + hf * 8, ox, oy, oz, Din,
                                            stride, padlo);
      v8bf a1 = load_a_frag<KS, DECONV, CI>(in, k0 + 16 + hf * 8, ox, oy, oz,
                                            Din, stride, padlo);
      v16bf a;
#pragma unroll
      for (int i = 0; i < 8; ++i) { a[i] = a0[i]; a[i + 8] = a1[i]; }
#pragma unroll
      for (int t = 0; t < NT; ++t) {
        const bf16_t* bp = brow0 + (size_t)t * 16 * Kpad + k0;
        v8bf b0 = *(const v8bf*)bp;
        v8bf b1 = *(const v8bf*)(bp + 16);
        v16bf b;
#pragma unroll
        for (int i = 0; i < 8; ++i) { b[i] = b0[i]; b[i + 8] = b1[i]; }
        acc[t] = __builtin_amdgcn_wmma_f32_16x16x32_bf16(
            false, a, false, b, (short)0, acc[t], false, false);
      }
    }
  } else {
    // ---------------- slow path (Ci==4, init layer only): LDS im2col ------
    __shared__ __align__(16) bf16_t As[16 * 40];
    const int rr = lane >> 1;
    const int cb = (lane & 1) << 4;
    const int ov2 = vbase + rr;
    const int oz2 = ov2 % Dout;
    const int oy2 = (ov2 / Dout) % Dout;
    const int ox2 = ov2 / (Dout * Dout);
    for (int k0 = 0; k0 < Kpad; k0 += 32) {
#pragma unroll
      for (int t = 0; t < 16; ++t) {
        int kk = k0 + cb + t;
        bf16_t bv = (bf16_t)0.0f;
        if (kk < Ktot) {
          int ci = kk % CI;
          int rem = kk / CI;
          int kw = rem % KS;
          int kh = (rem / KS) % KS;
          int kd = rem / (KS * KS);
          int ix = ox2 * stride + kd - padlo;
          int iy = oy2 * stride + kh - padlo;
          int iz = oz2 * stride + kw - padlo;
          bool ok = ((ix | iy | iz) >= 0) && (ix < Din) && (iy < Din) &&
                    (iz < Din);
          if (ok) bv = in[(((size_t)ix * Din + iy) * Din + iz) * CI + ci];
        }
        As[rr * 40 + cb + t] = bv;
      }
      __syncthreads();
      v8bf a0 = *(const v8bf*)&As[m * 40 + hf * 8];
      v8bf a1 = *(const v8bf*)&As[m * 40 + 16 + hf * 8];
      v16bf a;
#pragma unroll
      for (int i = 0; i < 8; ++i) { a[i] = a0[i]; a[i + 8] = a1[i]; }
#pragma unroll
      for (int t = 0; t < NT; ++t) {
        const bf16_t* bp = brow0 + (size_t)t * 16 * Kpad + k0;
        v8bf b0 = *(const v8bf*)bp;
        v8bf b1 = *(const v8bf*)(bp + 16);
        v16bf b;
#pragma unroll
        for (int i = 0; i < 8; ++i) { b[i] = b0[i]; b[i + 8] = b1[i]; }
        acc[t] = __builtin_amdgcn_wmma_f32_16x16x32_bf16(
            false, a, false, b, (short)0, acc[t], false, false);
      }
      __syncthreads();
    }
  }

  // ---- epilogue: D layout lane=N column, VGPR i = row 8*hf+i ----
#pragma unroll
  for (int t = 0; t < NT; ++t) {
    int nn = nb + t * 16 + m;
    float g = gamma ? gamma[nn] : 1.0f;
    float bt = beta ? beta[nn] : 0.0f;
#pragma unroll
    for (int i = 0; i < 8; ++i) {
      int mr = vbase + hf * 8 + i;
      float v = acc[t][i] * g + bt;
      if (res) v += (float)res[(size_t)mr * Co + nn];
      if (relu) v = v > 0.0f ? v : 0.0f;
      out[(size_t)mr * Co + nn] = (bf16_t)v;
    }
  }
}

// ------------------------------- concat ------------------------------------

__global__ void concat_kernel(const bf16_t* __restrict__ a,
                              const bf16_t* __restrict__ b,
                              bf16_t* __restrict__ o, int nv, int ca, int cb2) {
  int t = blockIdx.x * blockDim.x + threadIdx.x;
  int ct = ca + cb2;
  if (t >= nv * ct) return;
  int v = t / ct;
  int c = t - v * ct;
  o[t] = (c < ca) ? a[(size_t)v * ca + c] : b[(size_t)v * cb2 + (c - ca)];
}

// -------------------- trilinear devoxelize + L2 normalize ------------------

__global__ void devox_kernel(const float* __restrict__ xyz,
                             const bf16_t* __restrict__ vf,
                             float* __restrict__ out, int n) {
  int t = blockIdx.x * blockDim.x + threadIdx.x;
  if (t >= n * OUTC) return;
  int p = t / OUTC;
  int c = t - p * OUTC;
  float qx = xyz[p * 3 + 0] * (1.0f / VSZ) - 0.5f;
  float qy = xyz[p * 3 + 1] * (1.0f / VSZ) - 0.5f;
  float qz = xyz[p * 3 + 2] * (1.0f / VSZ) - 0.5f;
  float fx = floorf(qx), fy = floorf(qy), fz = floorf(qz);
  float wx = qx - fx, wy = qy - fy, wz = qz - fz;
  int x0 = (int)fx, y0 = (int)fy, z0 = (int)fz;
  float acc = 0.0f;
#pragma unroll
  for (int dx = 0; dx < 2; ++dx)
#pragma unroll
    for (int dy = 0; dy < 2; ++dy)
#pragma unroll
      for (int dz = 0; dz < 2; ++dz) {
        int cx = x0 + dx; cx = cx < 0 ? 0 : (cx > GRIDD - 1 ? GRIDD - 1 : cx);
        int cy = y0 + dy; cy = cy < 0 ? 0 : (cy > GRIDD - 1 ? GRIDD - 1 : cy);
        int cz = z0 + dz; cz = cz < 0 ? 0 : (cz > GRIDD - 1 ? GRIDD - 1 : cz);
        float wt = (dx ? wx : 1.0f - wx) * (dy ? wy : 1.0f - wy) *
                   (dz ? wz : 1.0f - wz);
        acc += (float)vf[(size_t)((cx * GRIDD + cy) * GRIDD + cz) * OUTC + c] * wt;
      }
  out[t] = acc;
}

__global__ __launch_bounds__(32) void norm_kernel(float* __restrict__ out) {
  int p = blockIdx.x;
  int l = threadIdx.x;
  float a0 = out[(size_t)p * OUTC + l];
  float a1 = out[(size_t)p * OUTC + 32 + l];
  float s = a0 * a0 + a1 * a1;
#pragma unroll
  for (int o = 16; o > 0; o >>= 1) s += __shfl_xor(s, o, 32);
  float inv = 1.0f / sqrtf(s);
  out[(size_t)p * OUTC + l] = a0 * inv;
  out[(size_t)p * OUTC + 32 + l] = a1 * inv;
}

// ------------------------------- host glue ---------------------------------

enum InIdx {
  IN_XYZ = 0, IN_FEAT,
  W_INIT_B, G_INIT_B, B_INIT_B,
  W_INIT_R1, G_INIT_R1, B_INIT_R1,
  W_INIT_R2, G_INIT_R2, B_INIT_R2,
  W_D1B, G_D1B, B_D1B,
  W_D1R1, G_D1R1, B_D1R1,
  W_D1R2, G_D1R2, B_D1R2,
  W_D2B, G_D2B, B_D2B,
  W_D2R1, G_D2R1, B_D2R1,
  W_D2R2, G_D2R2, B_D2R2,
  W_BT1, G_BT1, B_BT1,
  W_BT2, G_BT2, B_BT2,
  W_BTS, G_BTS, B_BTS,
  W_U1T, G_U1T, B_U1T,
  W_U1R1, G_U1R1, B_U1R1,
  W_U1R2, G_U1R2, B_U1R2,
  W_U2T, G_U2T, B_U2T,
  W_U2R1, G_U2R1, B_U2R1,
  W_U2R2, G_U2R2, B_U2R2,
  W_FINAL, B_FINAL
};

struct PW { bf16_t* p; int kpad; };

#define FP(i) ((const float*)d_in[i])

#define LCONV(KS, DEC, CI, NT, INP, W, GI, BI, RES, OUTP, Din, Dout, Co, STR, PAD, RELU) \
  conv3d_wmma_kernel<KS, DEC, CI, NT>                                                    \
      <<<dim3(((Dout) * (Dout) * (Dout)) / 16, (Co) / (16 * (NT))), dim3(32), 0,         \
         stream>>>(INP, (W).p, GI, BI, RES, OUTP, Din, Dout, Co, (W).kpad, STR, PAD, RELU)

extern "C" void kernel_launch(void* const* d_in, const int* in_sizes, int n_in,
                              void* d_out, int out_size, void* d_ws,
                              size_t ws_size, hipStream_t stream) {
  (void)n_in; (void)out_size; (void)ws_size;
  const float* xyz = (const float*)d_in[IN_XYZ];
  const float* feat = (const float*)d_in[IN_FEAT];
  const int npts = in_sizes[IN_XYZ] / 3;
  char* ws = (char*)d_ws;
  auto at = [&](size_t kib) -> char* { return ws + kib * 1024; };

  // workspace plan (KiB offsets; big buffers aliased once producers die)
  float*  sums = (float*)at(8192);    // 4 MiB
  float*  cnt  = (float*)at(12288);   // 1 MiB
  bf16_t* V0   = (bf16_t*)at(13312);  // 64^3 x 4
  bf16_t* X1   = (bf16_t*)at(16384);  // 64^3 x 16
  bf16_t* T1   = (bf16_t*)at(24576);
  bf16_t* S1   = (bf16_t*)at(32768);
  bf16_t* X2   = (bf16_t*)at(40960);  // 32^3 x 32
  bf16_t* T2   = (bf16_t*)at(43008);
  bf16_t* S2   = (bf16_t*)at(45056);
  bf16_t* X3   = (bf16_t*)at(47104);  // 16^3 x 64
  bf16_t* T3   = (bf16_t*)at(47616);
  bf16_t* S4   = (bf16_t*)at(48128);
  bf16_t* SK   = (bf16_t*)at(48640);  // 16^3 x 128
  bf16_t* T4   = (bf16_t*)at(49664);
  bf16_t* S4T  = (bf16_t*)at(50688);
  bf16_t* C1   = (bf16_t*)at(51712);  // 16^3 x 192
  bf16_t* X5   = (bf16_t*)at(53248);  // 32^3 x 64
  bf16_t* T5   = (bf16_t*)at(57344);
  bf16_t* U1   = (bf16_t*)at(61440);
  bf16_t* C2   = (bf16_t*)at(65536);  // 32^3 x 96
  bf16_t* X6   = (bf16_t*)at(16384);  // 64^3 x 48 (aliases X1/T1/S1, dead)
  bf16_t* T6   = (bf16_t*)at(71680);  // 64^3 x 48
  bf16_t* U2   = (bf16_t*)at(40960);  // 64^3 x 48 (aliases X2..U1, dead)
  bf16_t* VF   = (bf16_t*)at(65536);  // 64^3 x 64 (aliases C2/T6, dead)

  // ---- pack all conv weights fp32 -> bf16 [Co][Kpad] ----
  size_t woff = 0;
  auto pack = [&](int widx, int KS, int Ci, int Co) -> PW {
    int Ktot = KS * KS * KS * Ci;
    int Kpad = (Ktot + 31) & ~31;
    PW r{(bf16_t*)(ws + woff), Kpad};
    woff += ((size_t)Co * Kpad * sizeof(bf16_t) + 255) & ~(size_t)255;
    int total = Co * Kpad;
    pack_w_kernel<<<(total + 255) / 256, 256, 0, stream>>>(
        (const float*)d_in[widx], r.p, Ktot, Kpad, Co);
    return r;
  };
  PW w_init_b  = pack(W_INIT_B, 5, 4, 16);
  PW w_init_r1 = pack(W_INIT_R1, 3, 16, 16);
  PW w_init_r2 = pack(W_INIT_R2, 3, 16, 16);
  PW w_d1b  = pack(W_D1B, 3, 16, 32);
  PW w_d1r1 = pack(W_D1R1, 3, 32, 32);
  PW w_d1r2 = pack(W_D1R2, 3, 32, 32);
  PW w_d2b  = pack(W_D2B, 3, 32, 64);
  PW w_d2r1 = pack(W_D2R1, 3, 64, 64);
  PW w_d2r2 = pack(W_D2R2, 3, 64, 64);
  PW w_bt1  = pack(W_BT1, 3, 64, 128);
  PW w_bt2  = pack(W_BT2, 3, 128, 128);
  PW w_bts  = pack(W_BTS, 1, 64, 128);
  PW w_u1t  = pack(W_U1T, 3, 192, 64);
  PW w_u1r1 = pack(W_U1R1, 3, 64, 64);
  PW w_u1r2 = pack(W_U1R2, 3, 64, 64);
  PW w_u2t  = pack(W_U2T, 3, 96, 48);
  PW w_u2r1 = pack(W_U2R1, 3, 48, 48);
  PW w_u2r2 = pack(W_U2R2, 3, 48, 48);
  PW w_fin  = pack(W_FINAL, 1, 48, 64);

  // ---- voxelize: scatter-mean into 64^3 grid ----
  hipMemsetAsync(sums, 0, (size_t)5 * 1024 * 1024, stream);  // sums + cnt
  scatter_kernel<<<(npts + 255) / 256, 256, 0, stream>>>(xyz, feat, sums, cnt, npts);
  meanify_kernel<<<(GRIDD3 + 255) / 256, 256, 0, stream>>>(sums, cnt, V0);

  const bf16_t* nil = nullptr;

  // ---- encoder ----
  LCONV(5, false, 4, 1, V0, w_init_b, FP(G_INIT_B), FP(B_INIT_B), nil, X1, 64, 64, 16, 1, 2, 1);
  LCONV(3, false, 16, 1, X1, w_init_r1, FP(G_INIT_R1), FP(B_INIT_R1), nil, T1, 64, 64, 16, 1, 1, 1);
  LCONV(3, false, 16, 1, T1, w_init_r2, FP(G_INIT_R2), FP(B_INIT_R2), X1, S1, 64, 64, 16, 1, 1, 1);

  LCONV(3, false, 16, 2, S1, w_d1b, FP(G_D1B), FP(B_D1B), nil, X2, 64, 32, 32, 2, 0, 1);
  LCONV(3, false, 32, 2, X2, w_d1r1, FP(G_D1R1), FP(B_D1R1), nil, T2, 32, 32, 32, 1, 1, 1);
  LCONV(3, false, 32, 2, T2, w_d1r2, FP(G_D1R2), FP(B_D1R2), X2, S2, 32, 32, 32, 1, 1, 1);

  LCONV(3, false, 32, 2, S2, w_d2b, FP(G_D2B), FP(B_D2B), nil, X3, 32, 16, 64, 2, 0, 1);
  LCONV(3, false, 64, 2, X3, w_d2r1, FP(G_D2R1), FP(B_D2R1), nil, T3, 16, 16, 64, 1, 1, 1);
  LCONV(3, false, 64, 2, T3, w_d2r2, FP(G_D2R2), FP(B_D2R2), X3, S4, 16, 16, 64, 1, 1, 1);

  // ---- bottleneck (channel-expanding skip) ----
  LCONV(1, false, 64, 2, S4, w_bts, FP(G_BTS), FP(B_BTS), nil, SK, 16, 16, 128, 1, 0, 0);
  LCONV(3, false, 64, 2, S4, w_bt1, FP(G_BT1), FP(B_BT1), nil, T4, 16, 16, 128, 1, 1, 1);
  LCONV(3, false, 128, 2, T4, w_bt2, FP(G_BT2), FP(B_BT2), SK, S4T, 16, 16, 128, 1, 1, 1);

  // ---- decoder stage 1 ----
  {
    int tot = 4096 * 192;
    concat_kernel<<<(tot + 255) / 256, 256, 0, stream>>>(S4T, S4, C1, 4096, 128, 64);
  }
  LCONV(3, true, 192, 2, C1, w_u1t, FP(G_U1T), FP(B_U1T), nil, X5, 16, 32, 64, 2, 2, 1);
  LCONV(3, false, 64, 2, X5, w_u1r1, FP(G_U1R1), FP(B_U1R1), nil, T5, 32, 32, 64, 1, 1, 1);
  LCONV(3, false, 64, 2, T5, w_u1r2, FP(G_U1R2), FP(B_U1R2), X5, U1, 32, 32, 64, 1, 1, 1);

  // ---- decoder stage 2 ----
  {
    int tot = 32768 * 96;
    concat_kernel<<<(tot + 255) / 256, 256, 0, stream>>>(U1, S2, C2, 32768, 64, 32);
  }
  LCONV(3, true, 96, 3, C2, w_u2t, FP(G_U2T), FP(B_U2T), nil, X6, 32, 64, 48, 2, 2, 1);
  LCONV(3, false, 48, 3, X6, w_u2r1, FP(G_U2R1), FP(B_U2R1), nil, T6, 64, 64, 48, 1, 1, 1);
  LCONV(3, false, 48, 3, T6, w_u2r2, FP(G_U2R2), FP(B_U2R2), X6, U2, 64, 64, 48, 1, 1, 1);

  // ---- final 1x1 conv (gamma=1, beta=bias, no relu) ----
  LCONV(1, false, 48, 2, U2, w_fin, (const float*)nullptr, FP(B_FINAL), nil, VF, 64, 64, 64, 1, 0, 0);

  // ---- devoxelize + normalize ----
  devox_kernel<<<((npts * OUTC) + 255) / 256, 256, 0, stream>>>(xyz, VF, (float*)d_out, npts);
  norm_kernel<<<npts, 32, 0, stream>>>((float*)d_out);
}
